// BigBirdMHAttention_69827578299063
// MI455X (gfx1250) — matrix operand order repaired and checked by reading, not compiled, not run on hardware
//
#include <hip/hip_runtime.h>

// ---------------------------------------------------------------------------
// BigBird multi-head attention for MI455X (gfx1250, wave32, WMMA bf16).
// All GEMM-like contractions go through v_wmma_f32_16x16x32_bf16.
// V-tile staging uses the Tensor Data Mover (tensor_load_to_lds, 6-arg form)
// and Q-tile staging uses GLOBAL_LOAD_ASYNC_TO_LDS_B128.
// ---------------------------------------------------------------------------

typedef __bf16 bf16_t;
typedef __attribute__((ext_vector_type(16))) __bf16 v16bf;
typedef __attribute__((ext_vector_type(8)))  __bf16 v8bf;
typedef __attribute__((ext_vector_type(8)))  float  v8f;
typedef unsigned int v4u __attribute__((ext_vector_type(4)));
typedef int v8i __attribute__((ext_vector_type(8)));
typedef int v4i __attribute__((ext_vector_type(4)));

#define BATCH  2
#define SEQ    4096
#define DIM    1024
#define NHEADS 16
#define DH     64
#define NB     128        // SEQ / 32
#define NEGVAL -1.0e9f

#if defined(__AMDGCN__) && __has_builtin(__builtin_amdgcn_tensor_load_to_lds)
#define HAVE_TDM 1
#else
#define HAVE_TDM 0
#endif

#if defined(__AMDGCN__) && \
    __has_builtin(__builtin_amdgcn_global_load_async_to_lds_b128)
#define HAVE_ASYNC 1
#else
#define HAVE_ASYNC 0
#endif

#if defined(__AMDGCN__) && __has_builtin(__builtin_amdgcn_s_wait_asynccnt)
#define WAIT_ASYNC() __builtin_amdgcn_s_wait_asynccnt(0)
#else
#define WAIT_ASYNC() ((void)0)
#endif

// ---------------------------------------------------------------------------
// WMMA helpers (layouts per CDNA5 ISA 7.12.2)
// ---------------------------------------------------------------------------
__device__ inline v8f wmma_bf16(v16bf a, v16bf b, v8f c) {
  return __builtin_amdgcn_wmma_f32_16x16x32_bf16(
      /*neg_a=*/false, a, /*neg_b=*/false, b,
      /*c_mod=*/(short)0, c, /*reuse_a=*/false, /*reuse_b=*/false);
}

// A-matrix 16x32 (MxK), row-major source with leading dim ld.
__device__ inline v16bf frag_a_bf16(const bf16_t* base, int ld, int lane) {
  int row = lane & 15;
  int kb  = (lane < 16) ? 0 : 8;
  const bf16_t* p = base + row * ld + kb;
  v8bf lo = *(const v8bf*)p;
  v8bf hi = *(const v8bf*)(p + 16);
  v16bf a;
#pragma unroll
  for (int i = 0; i < 8; ++i) { a[i] = lo[i]; a[i + 8] = hi[i]; }
  return a;
}

// Same A layout, fp32 source (converted on the fly).
__device__ inline v16bf frag_a_f32(const float* base, int ld, int lane) {
  int row = lane & 15;
  int kb  = (lane < 16) ? 0 : 8;
  const float* p = base + row * ld + kb;
  v16bf a;
#pragma unroll
  for (int i = 0; i < 8; ++i) {
    a[i]     = (bf16_t)p[i];
    a[i + 8] = (bf16_t)p[i + 16];
  }
  return a;
}

// B-matrix 32x16 (KxN): lane holds col=lane&15 and K values kb..kb+7 and
// kb+16..kb+23 with kb=(lane<16 ? 0 : 8)+kofs. rowp = this lane's column stored
// K-contiguous (i.e. a Bt row).
__device__ inline v16bf frag_b_rowptr(const bf16_t* rowp, int kofs, int lane) {
  int kb = ((lane < 16) ? 0 : 8) + kofs;
  v8bf lo = *(const v8bf*)(rowp + kb);
  v8bf hi = *(const v8bf*)(rowp + kb + 16);
  v16bf b;
#pragma unroll
  for (int i = 0; i < 8; ++i) { b[i] = lo[i]; b[i + 8] = hi[i]; }
  return b;
}

// ---------------------------------------------------------------------------
// Tensor Data Mover: contiguous 4 KB copy (512 x 8B elements) to LDS.
// D# per CDNA5 ISA 8.3/8.4: count=1, type=2, data_size=3(8B), 1-row tile.
// 6-arg builtin form: (g0, g1, g2, g3, g2b, cpol) with zero-filled groups 2/3.
// ---------------------------------------------------------------------------
#if HAVE_TDM
__device__ inline void tdm_copy_4kb(const void* gsrc, unsigned lds_off) {
  unsigned long long ga = (unsigned long long)(size_t)gsrc;
  v4u g0;
  g0[0] = 1u;                                   // count=1 (valid user D#)
  g0[1] = lds_off;                              // lds_addr (bytes)
  g0[2] = (unsigned)ga;                         // global_addr[31:0]
  g0[3] = (unsigned)((ga >> 32) & 0x01FFFFFFu)  // global_addr[56:32]
          | (2u << 30);                         // type = 2 ("image")
  v8i g1;
  const int elems = 512;                        // 4 KB / 8B
  g1[0] = (3 << 16);            // workgroup_mask=0, data_size=3 (8 bytes)
  g1[1] = (elems & 0xFFFF) << 16;               // tensor_dim0[15:0]
  g1[2] = (elems >> 16) | (1 << 16);            // tensor_dim0 hi | tensor_dim1=1
  g1[3] = (elems << 16);                        // tensor_dim1 hi=0 | tile_dim0
  g1[4] = 1;                                    // tile_dim1=1, tile_dim2=0
  g1[5] = elems;                                // tensor_dim0_stride lo
  g1[6] = 0;
  g1[7] = 0;
  v4i z4 = {0, 0, 0, 0};
  v8i z8 = {0, 0, 0, 0, 0, 0, 0, 0};
  __builtin_amdgcn_tensor_load_to_lds(g0, g1, z4, z4, z8, 0);
}
#endif

// ---------------------------------------------------------------------------
// fp32 -> bf16 conversion
// ---------------------------------------------------------------------------
__global__ __launch_bounds__(256) void f32_to_bf16_kernel(
    const float* __restrict__ in, bf16_t* __restrict__ out, int n) {
  int i = blockIdx.x * blockDim.x + threadIdx.x;
  int stride = gridDim.x * blockDim.x;
  for (; i < n; i += stride) out[i] = (bf16_t)in[i];
}

// ---------------------------------------------------------------------------
// GEMM: out[m][n] = (sum_k A[m][k]*Bt[n][k] + bias[n]) * scaleOut
// mode 0: fp32 out row-major MxN; mode 2: bf16 out permuted to [B,H,S,Dh].
// Workgroup: 8 waves -> 128(M) x 64(N) tile; each wave a 32x32 tile.
// ---------------------------------------------------------------------------
__global__ __launch_bounds__(256) void gemm_bf16_kernel(
    const bf16_t* __restrict__ A, const bf16_t* __restrict__ Bt,
    const float* __restrict__ bias, void* __restrict__ out,
    int M, int N, int K, int mode, float scaleOut) {
  const int lane = threadIdx.x & 31;
  const int wave = threadIdx.x >> 5;
  const int m0 = blockIdx.x * 128 + (wave & 3) * 32;
  const int n0 = blockIdx.y * 64 + (wave >> 2) * 32;
  const int colBase = lane & 15;
  const int rowAdd  = (lane < 16) ? 0 : 8;

  v8f acc[2][2] = {};
  for (int k = 0; k < K; k += 32) {
    v16bf a0 = frag_a_bf16(A + (size_t)m0 * K + k, K, lane);
    v16bf a1 = frag_a_bf16(A + (size_t)(m0 + 16) * K + k, K, lane);
    v16bf b0 = frag_b_rowptr(Bt + (size_t)(n0 + colBase) * K, k, lane);
    v16bf b1 = frag_b_rowptr(Bt + (size_t)(n0 + 16 + colBase) * K, k, lane);
    acc[0][0] = wmma_bf16(a0, b0, acc[0][0]);
    acc[0][1] = wmma_bf16(a0, b1, acc[0][1]);
    acc[1][0] = wmma_bf16(a1, b0, acc[1][0]);
    acc[1][1] = wmma_bf16(a1, b1, acc[1][1]);
  }

#pragma unroll
  for (int mt = 0; mt < 2; ++mt)
#pragma unroll
    for (int nt = 0; nt < 2; ++nt)
#pragma unroll
      for (int j = 0; j < 8; ++j) {
        int m = m0 + mt * 16 + rowAdd + j;
        int n = n0 + nt * 16 + colBase;
        float v = (acc[mt][nt][j] + bias[n]) * scaleOut;
        if (mode == 0) {
          ((float*)out)[(size_t)m * N + n] = v;
        } else {  // mode 2: [B,H,S,Dh]
          int b = m >> 12, s = m & (SEQ - 1);
          int h = n >> 6, dh = n & (DH - 1);
          ((bf16_t*)out)[(((size_t)(b * NHEADS + h)) * SEQ + s) * DH + dh] =
              (bf16_t)v;
        }
      }
}

// ---------------------------------------------------------------------------
// Key mapping for windowed attention: 256 keys = 5 window blocks (n-2..n+2,
// clipped, globals excluded) + 3 global blocks {1,3,5}.
// ---------------------------------------------------------------------------
__device__ inline void keymap(int n, int key, const unsigned char* maskrow,
                              int& pos, bool& valid) {
  int blk = key >> 5, kk = key & 31;
  if (blk < 5) {
    int nbr = n - 2 + blk;
    bool ok = (nbr >= 0) && (nbr < NB) && !(nbr == 1 || nbr == 3 || nbr == 5);
    int nbc = nbr < 0 ? 0 : (nbr > NB - 1 ? NB - 1 : nbr);
    pos = nbc * 32 + kk;
    valid = ok && (maskrow[pos] != 0);
  } else {
    int g = (blk == 5) ? 1 : ((blk == 6) ? 3 : 5);
    pos = g * 32 + kk;
    valid = (maskrow[pos] != 0);
  }
}

// ---------------------------------------------------------------------------
// Windowed + global-column attention for one (b,h,block).
// ---------------------------------------------------------------------------
__global__ __launch_bounds__(256) void bigbird_window_kernel(
    const bf16_t* __restrict__ q, const bf16_t* __restrict__ k,
    const bf16_t* __restrict__ v, const unsigned char* __restrict__ mask,
    bf16_t* __restrict__ o) {
  const int n = blockIdx.x & (NB - 1);
  const int h = (blockIdx.x >> 7) & (NHEADS - 1);
  const int b = blockIdx.x >> 11;
  if (n == 1 || n == 3 || n == 5) return;

  __shared__ float  sS[32 * 256];    // scores -> probabilities (32 KB)
  __shared__ bf16_t sVt[DH * 256];   // transposed V tile [dh][key] (32 KB)
  __shared__ bf16_t sQ[32 * DH];     // Q tile (4 KB)
#if HAVE_TDM
  __shared__ bf16_t sVraw[256 * DH]; // row-major V tile, TDM dest (32 KB)
#endif

  const int tid = threadIdx.x, lane = tid & 31, wave = tid >> 5;
  const int colBase = lane & 15;
  const int rowAdd  = (lane < 16) ? 0 : 8;
  const unsigned char* maskrow = mask + (size_t)b * SEQ;
  const size_t bh = (size_t)(b * NHEADS + h) * SEQ;
  const bf16_t* qsrc = q + (bh + n * 32) * DH;  // pre-scaled by 1/sqrt(Dh)

  // ---- stage Q tile into LDS (async copy if available) ----
#if HAVE_ASYNC
  __builtin_amdgcn_global_load_async_to_lds_b128(
      (v4i*)(qsrc + tid * 8), (v4i*)(sQ + tid * 8), 0, 0);
#else
  *(v8bf*)(sQ + tid * 8) = *(const v8bf*)(qsrc + tid * 8);
#endif

  // ---- stage V (TDM DMA per 4 KB key-block if available) ----
#if HAVE_TDM
  if (wave == 0) {
#pragma unroll
    for (int j = 0; j < 8; ++j) {
      int pos; bool valid;
      keymap(n, j * 32, maskrow, pos, valid);
      tdm_copy_4kb(v + (bh + pos) * DH,
                   (unsigned)(size_t)(void*)(sVraw + j * 32 * DH));
    }
    __builtin_amdgcn_s_wait_tensorcnt(0);
  }
  WAIT_ASYNC();
  __syncthreads();
  // Transpose sVraw -> sVt (thread t = key t).
#pragma unroll
  for (int c = 0; c < 8; ++c) {
    v8bf chunk = *(const v8bf*)(sVraw + tid * DH + c * 8);
#pragma unroll
    for (int i = 0; i < 8; ++i) sVt[(c * 8 + i) * 256 + tid] = chunk[i];
  }
#else
  {
    int pos; bool valid;
    keymap(n, tid, maskrow, pos, valid);
    const bf16_t* vrow = v + (bh + pos) * DH;
#pragma unroll
    for (int c = 0; c < 8; ++c) {
      v8bf chunk = *(const v8bf*)(vrow + c * 8);
#pragma unroll
      for (int i = 0; i < 8; ++i) sVt[(c * 8 + i) * 256 + tid] = chunk[i];
    }
  }
  WAIT_ASYNC();
  __syncthreads();
#endif

  // ---- scores: wave handles key columns [wave*32, wave*32+32) ----
  {
    v8f c[2][2] = {};
    int pos0, pos1; bool val0, val1;
    keymap(n, wave * 32 + colBase, maskrow, pos0, val0);
    keymap(n, wave * 32 + 16 + colBase, maskrow, pos1, val1);
    const bf16_t* kr0 = k + (bh + pos0) * DH;
    const bf16_t* kr1 = k + (bh + pos1) * DH;
#pragma unroll
    for (int kd = 0; kd < DH; kd += 32) {
      v16bf a0 = frag_a_bf16(sQ + kd, DH, lane);
      v16bf a1 = frag_a_bf16(sQ + 16 * DH + kd, DH, lane);
      v16bf b0 = frag_b_rowptr(kr0, kd, lane);
      v16bf b1 = frag_b_rowptr(kr1, kd, lane);
      c[0][0] = wmma_bf16(a0, b0, c[0][0]);
      c[0][1] = wmma_bf16(a0, b1, c[0][1]);
      c[1][0] = wmma_bf16(a1, b0, c[1][0]);
      c[1][1] = wmma_bf16(a1, b1, c[1][1]);
    }
#pragma unroll
    for (int mt = 0; mt < 2; ++mt)
#pragma unroll
      for (int nt = 0; nt < 2; ++nt) {
        bool valid = nt ? val1 : val0;
#pragma unroll
        for (int j = 0; j < 8; ++j) {
          int row = mt * 16 + rowAdd + j;
          int col = wave * 32 + nt * 16 + colBase;
          sS[row * 256 + col] = valid ? c[mt][nt][j] : NEGVAL;
        }
      }
  }
  __syncthreads();

  // ---- softmax: wave handles 4 rows; 8 lanes per row, 32 cols per lane ----
  {
    int r = wave * 4 + (lane >> 3);
    float* rowp = sS + r * 256 + (lane & 7) * 32;
    float mx = NEGVAL;
#pragma unroll
    for (int i = 0; i < 32; ++i) mx = fmaxf(mx, rowp[i]);
#pragma unroll
    for (int off = 1; off < 8; off <<= 1) mx = fmaxf(mx, __shfl_xor(mx, off, 8));
    float sum = 0.f;
#pragma unroll
    for (int i = 0; i < 32; ++i) {
      float e = __expf(rowp[i] - mx);
      rowp[i] = e;
      sum += e;
    }
#pragma unroll
    for (int off = 1; off < 8; off <<= 1) sum += __shfl_xor(sum, off, 8);
    float inv = 1.f / sum;
#pragma unroll
    for (int i = 0; i < 32; ++i) rowp[i] *= inv;
  }
  __syncthreads();

  // ---- PV: wave computes one 16x16 output tile ----
  {
    int mt = wave & 1, ntd = wave >> 1;
    v8f oc = (v8f){};
#pragma unroll
    for (int kt = 0; kt < 256; kt += 32) {
      v16bf pa = frag_a_f32(sS + mt * 16 * 256 + kt, 256, lane);
      const bf16_t* vb = sVt + (ntd * 16 + colBase) * 256 + kt;
      v16bf bf = frag_b_rowptr(vb, 0, lane);
      oc = wmma_bf16(pa, bf, oc);
    }
#pragma unroll
    for (int j = 0; j < 8; ++j) {
      int row = mt * 16 + rowAdd + j;
      int s = n * 32 + row;
      int d = h * DH + ntd * 16 + colBase;
      o[((size_t)b * SEQ + s) * DIM + d] = (bf16_t)oc[j];
    }
  }
}

// ---------------------------------------------------------------------------
// Dense (flash-style) attention for the 96 global q rows per (b,h) over all
// 4096 keys. 6 active waves, each owns 16 q rows; streaming 32-key tiles.
// ---------------------------------------------------------------------------
__global__ __launch_bounds__(256) void bigbird_dense_kernel(
    const bf16_t* __restrict__ q, const bf16_t* __restrict__ k,
    const bf16_t* __restrict__ v, const unsigned char* __restrict__ mask,
    bf16_t* __restrict__ o) {
  const int h = blockIdx.x & (NHEADS - 1);
  const int b = blockIdx.x >> 4;
  const int tid = threadIdx.x, lane = tid & 31, wave = tid >> 5;
  const int colBase = lane & 15;
  const int rowAdd  = (lane < 16) ? 0 : 8;

  __shared__ bf16_t sVt[DH * 32];     // V^T tile [dh][key]
  __shared__ float  sP[6 * 16 * 32];  // per-wave probabilities
  __shared__ float  sBad[32];         // 1.0 = masked-out key

  const size_t bh = (size_t)(b * NHEADS + h) * SEQ;
  const int gsel = wave >> 1;  // 0,1,2 -> global blocks 1,3,5
  const int gblk = (gsel == 0) ? 1 : (gsel == 1 ? 3 : 5);
  const bf16_t* qbase = q + (bh + gblk * 32 + (wave & 1) * 16) * DH;

  // Loop-invariant Q fragments (hoisted: barrier in the loop blocks LICM).
  v16bf aq0 = {}, aq1 = {};
  if (wave < 6) {
    aq0 = frag_a_bf16(qbase, DH, lane);
    aq1 = frag_a_bf16(qbase + 32, DH, lane);
  }

  float rm[8], rs[8];
  v8f acc[4];
#pragma unroll
  for (int j = 0; j < 8; ++j) { rm[j] = NEGVAL; rs[j] = 0.f; }
#pragma unroll
  for (int t = 0; t < 4; ++t) acc[t] = (v8f){};

  for (int kt0 = 0; kt0 < SEQ; kt0 += 32) {
    __syncthreads();  // previous iteration finished with sVt
    {
      int key = tid >> 3;
      int dc = (tid & 7) * 8;
      const bf16_t* vrow = v + (bh + kt0 + key) * DH + dc;
      v8bf chunk = *(const v8bf*)vrow;
#pragma unroll
      for (int i = 0; i < 8; ++i) sVt[(dc + i) * 32 + key] = chunk[i];
      if (tid < 32) sBad[tid] = mask[(size_t)b * SEQ + kt0 + tid] ? 0.f : 1.f;
    }
    __syncthreads();

    if (wave < 6) {
      v8f sc[2] = {(v8f){}, (v8f){}};
#pragma unroll
      for (int nt = 0; nt < 2; ++nt) {
        const bf16_t* kr = k + (bh + kt0 + nt * 16 + colBase) * DH;
        v16bf b0 = frag_b_rowptr(kr, 0, lane);
        v16bf b1 = frag_b_rowptr(kr, 32, lane);
        sc[nt] = wmma_bf16(aq0, b0, sc[nt]);
        sc[nt] = wmma_bf16(aq1, b1, sc[nt]);
      }
      bool bad0 = sBad[colBase] != 0.f;
      bool bad1 = sBad[16 + colBase] != 0.f;
#pragma unroll
      for (int j = 0; j < 8; ++j) {
        float v0 = bad0 ? NEGVAL : sc[0][j];
        float v1 = bad1 ? NEGVAL : sc[1][j];
        float tm = fmaxf(v0, v1);
#pragma unroll
        for (int off = 1; off < 16; off <<= 1)
          tm = fmaxf(tm, __shfl_xor(tm, off, 16));
        float nm = fmaxf(rm[j], tm);
        float corr = __expf(rm[j] - nm);
        float p0 = __expf(v0 - nm);
        float p1 = __expf(v1 - nm);
        float ps = p0 + p1;
#pragma unroll
        for (int off = 1; off < 16; off <<= 1) ps += __shfl_xor(ps, off, 16);
        rs[j] = rs[j] * corr + ps;
        rm[j] = nm;
#pragma unroll
        for (int t = 0; t < 4; ++t) acc[t][j] *= corr;
        int row = rowAdd + j;
        sP[(wave * 16 + row) * 32 + colBase] = p0;
        sP[(wave * 16 + row) * 32 + 16 + colBase] = p1;
      }
      // PV accumulate (same-wave LDS DS ops are in-order).
      v16bf pa = frag_a_f32(sP + wave * 16 * 32, 32, lane);
#pragma unroll
      for (int nt = 0; nt < 4; ++nt) {
        const bf16_t* vb = sVt + (nt * 16 + colBase) * 32;
        v16bf bf = frag_b_rowptr(vb, 0, lane);
        acc[nt] = wmma_bf16(pa, bf, acc[nt]);
      }
    }
  }

  if (wave < 6) {
#pragma unroll
    for (int nt = 0; nt < 4; ++nt)
#pragma unroll
      for (int j = 0; j < 8; ++j) {
        int row = rowAdd + j;
        int s = gblk * 32 + (wave & 1) * 16 + row;
        int d = h * DH + nt * 16 + colBase;
        o[((size_t)b * SEQ + s) * DIM + d] = (bf16_t)(acc[nt][j] / rs[j]);
      }
  }
}

// ---------------------------------------------------------------------------
// Launch
// ---------------------------------------------------------------------------
extern "C" void kernel_launch(void* const* d_in, const int* in_sizes, int n_in,
                              void* d_out, int out_size, void* d_ws,
                              size_t ws_size, hipStream_t stream) {
  (void)in_sizes; (void)n_in; (void)out_size; (void)ws_size;
  const float* x = (const float*)d_in[0];
  const unsigned char* mask = (const unsigned char*)d_in[1];  // jax bool
  const float* Wq = (const float*)d_in[2];
  const float* bq = (const float*)d_in[3];
  const float* Wk = (const float*)d_in[4];
  const float* bk = (const float*)d_in[5];
  const float* Wv = (const float*)d_in[6];
  const float* bv = (const float*)d_in[7];
  const float* Wo = (const float*)d_in[8];
  const float* bo = (const float*)d_in[9];

  const size_t E = (size_t)BATCH * SEQ * DIM;  // 8388608
  const size_t WE = (size_t)DIM * DIM;         // 1048576
  bf16_t* x16 = (bf16_t*)d_ws;
  bf16_t* q16 = x16 + E;
  bf16_t* k16 = q16 + E;
  bf16_t* v16 = k16 + E;
  bf16_t* o16 = v16 + E;
  bf16_t* wq16 = o16 + E;
  bf16_t* wk16 = wq16 + WE;
  bf16_t* wv16 = wk16 + WE;
  bf16_t* wo16 = wv16 + WE;

  f32_to_bf16_kernel<<<2048, 256, 0, stream>>>(x, x16, (int)E);
  f32_to_bf16_kernel<<<512, 256, 0, stream>>>(Wq, wq16, (int)WE);
  f32_to_bf16_kernel<<<512, 256, 0, stream>>>(Wk, wk16, (int)WE);
  f32_to_bf16_kernel<<<512, 256, 0, stream>>>(Wv, wv16, (int)WE);
  f32_to_bf16_kernel<<<512, 256, 0, stream>>>(Wo, wo16, (int)WE);

  dim3 gg(64, 16);  // M/128 x N/64
  // Q is pre-scaled by 1/sqrt(Dh) = 0.125
  gemm_bf16_kernel<<<gg, 256, 0, stream>>>(x16, wq16, bq, q16, 8192, 1024,
                                           1024, 2, 0.125f);
  gemm_bf16_kernel<<<gg, 256, 0, stream>>>(x16, wk16, bk, k16, 8192, 1024,
                                           1024, 2, 1.0f);
  gemm_bf16_kernel<<<gg, 256, 0, stream>>>(x16, wv16, bv, v16, 8192, 1024,
                                           1024, 2, 1.0f);

  bigbird_window_kernel<<<BATCH * NHEADS * NB, 256, 0, stream>>>(
      q16, k16, v16, mask, o16);
  bigbird_dense_kernel<<<BATCH * NHEADS, 256, 0, stream>>>(q16, k16, v16, mask,
                                                           o16);

  gemm_bf16_kernel<<<gg, 256, 0, stream>>>(o16, wo16, bo, d_out, 8192, 1024,
                                           1024, 0, 1.0f);
}
